// gpnet2_1_4_58763742544053
// MI455X (gfx1250) — matrix operand order repaired and checked by reading, not compiled, not run on hardware
//
#include <hip/hip_runtime.h>
#include <stdint.h>

typedef __attribute__((ext_vector_type(16))) _Float16 v16h;
typedef __attribute__((ext_vector_type(8)))  _Float16 v8h;
typedef __attribute__((ext_vector_type(8)))  float    v8f;

#define WAVES_PER_BLOCK 8
#define THREADS (WAVES_PER_BLOCK * 32)
#define HSTRIDE 72   // padded hT row stride in halves (bank-conflict avoidance)

__device__ __forceinline__ v8f wmma_f16(v16h a, v16h b, v8f c) {
    // 8 args: (neg_a, A, neg_b, B, c_mod, C, reuse_a, reuse_b)
    return __builtin_amdgcn_wmma_f32_16x16x32_f16(false, a, false, b, (short)0, c,
                                                  false, false);
}

// Hardware tanh: gfx1250 V_TANH_F32 (single TRANS instruction).
__device__ __forceinline__ float fast_tanh(float x) {
#if __has_builtin(__builtin_amdgcn_tanhf)
    return __builtin_amdgcn_tanhf(x);
#elif __has_builtin(__builtin_amdgcn_tanh_f32)
    return __builtin_amdgcn_tanh_f32(x);
#else
    float r;
    asm("v_tanh_f32 %0, %1\n\tv_nop" : "=v"(r) : "v"(x));
    return r;
#endif
}

__device__ __forceinline__ float fast_sigmoid(float y) {
    const float t = __builtin_amdgcn_exp2f(y * -1.442695040888963f); // exp(-y)
    return __builtin_amdgcn_rcpf(1.0f + t);
}

__device__ __forceinline__ void store_tanh(_Float16* p, v8f a) {
    v8h o;
    #pragma unroll
    for (int r = 0; r < 8; ++r) o[r] = (_Float16)fast_tanh(a[r]);
    *(v8h*)p = o;
}

__global__ __launch_bounds__(THREADS)
void mlp_forward_kernel(const float* __restrict__ x,
                        const float* __restrict__ W1, const float* __restrict__ B1,
                        const float* __restrict__ W2, const float* __restrict__ B2,
                        const float* __restrict__ W3, const float* __restrict__ B3,
                        const float* __restrict__ W4, const float* __restrict__ B4,
                        const float* __restrict__ W5, const float* __restrict__ B5,
                        const float* __restrict__ W6, const float* __restrict__ B6,
                        const float* __restrict__ W7, const float* __restrict__ B7,
                        const float* __restrict__ W8, const float* __restrict__ B8,
                        const float* __restrict__ scale,
                        float* __restrict__ out, int Ntot)
{
    // Layer geometry; mid layers 1..6 padded to a UNIFORM 4 m-tiles x 2 k-chunks
    // (L7's mt=3 tile is all zeros -> exact zero padding in hT for L8).
    const int FIN [8] = {2, 50, 50, 50, 50, 50, 50, 40};
    const int FOUT[8] = {50, 50, 50, 50, 50, 50, 40, 1};
    const int MT  [8] = {4, 4, 4, 4, 4, 4, 4, 1};
    const int KC  [8] = {1, 2, 2, 2, 2, 2, 2, 2};
    const int LBASE[8]= {0, 2048, 6144, 10240, 14336, 18432, 22528, 26624}; // halves

    __shared__ __align__(32) _Float16 wfrag[27648];                   // 54 KB fragments
    __shared__ __align__(32) float    biasS[512];                     // 8 x 64 padded
    __shared__ __align__(32) _Float16 hT[WAVES_PER_BLOCK * 32 * HSTRIDE]; // 2 tiles/wave

    const float* Ws[8] = {W1, W2, W3, W4, W5, W6, W7, W8};
    const float* Bs[8] = {B1, B2, B3, B4, B5, B6, B7, B8};

    // ---- One-time per block: swizzle weights into A-fragment layout in LDS ----
    // A 16x32 f16 (this code's contiguous k order, same as B): lane l (g=l>>4, m=l&15) holds row M=m,
// halves j=0..15 -> K = 32c + 16g + j.
    for (int l = 0; l < 8; ++l) {
        const int total = MT[l] * KC[l] * 512;   // halves this layer
        const float* W = Ws[l];
        const int fo = FOUT[l], fi = FIN[l], kc = KC[l], base = LBASE[l];
        for (int i = threadIdx.x; i < total; i += THREADS) {
            const int j    = i & 15;
            const int ln   = (i >> 4) & 31;
            const int frag = i >> 9;            // = mt*kc + c
            const int c    = frag % kc;
            const int mt   = frag / kc;
            const int gg   = ln >> 4;
            const int mm   = ln & 15;
            const int K    = 32 * c + 16 * gg + j;
            const int outf = 16 * mt + mm;
            const float v  = (K < fi && outf < fo) ? W[K * fo + outf] : 0.0f;
            wfrag[base + i] = (_Float16)v;
        }
    }
    for (int i = threadIdx.x; i < 512; i += THREADS) {
        const int l = i >> 6, cc = i & 63;
        biasS[i] = (cc < FOUT[l]) ? Bs[l][cc] : 0.0f;
    }
    __syncthreads();

    const int lane = threadIdx.x & 31;
    const int wave = threadIdx.x >> 5;
    const int g    = lane >> 4;      // half-wave group
    const int n    = lane & 15;      // batch column within tile
    _Float16* hp0 = hT + wave * (32 * HSTRIDE) + n * HSTRIDE;      // batch tile 0
    _Float16* hp1 = hp0 + 16 * HSTRIDE;                            // batch tile 1

    const float   sc     = scale[0];
    const int64_t tiles  = ((int64_t)Ntot + 31) >> 5;              // 32 rows per wave-iter
    const int64_t gw     = (int64_t)blockIdx.x * WAVES_PER_BLOCK + wave;
    const int64_t stride = (int64_t)gridDim.x * WAVES_PER_BLOCK;

    for (int64_t t = gw; t < tiles; t += stride) {
        const int64_t r0 = t * 32 + n;
        const int64_t r1 = r0 + 16;

        // ---- Layer 1 (2 -> 50): B fragments built directly from x ----
        v16h bx0 = {}, bx1 = {};
        if (g == 0) {
            if (r0 < (int64_t)Ntot) {
                const float2 xv = ((const float2*)x)[r0];
                bx0[0] = (_Float16)xv.x;  bx0[1] = (_Float16)xv.y;
            }
            if (r1 < (int64_t)Ntot) {
                const float2 xv = ((const float2*)x)[r1];
                bx1[0] = (_Float16)xv.x;  bx1[1] = (_Float16)xv.y;
            }
        }
        #pragma unroll
        for (int mt = 0; mt < 4; ++mt) {
            const v8f  bias = *(const v8f*)(biasS + 16 * mt + 8 * g);
            const v16h a0   = *(const v16h*)(wfrag + mt * 512 + lane * 16);
            const v8f acc0 = wmma_f16(a0, bx0, bias);
            const v8f acc1 = wmma_f16(a0, bx1, bias);
            store_tanh(hp0 + 16 * mt + 8 * g, acc0);
            store_tanh(hp1 + 16 * mt + 8 * g, acc1);
        }

        // ---- Layers 2..7 (uniform 4 m-tiles x 2 k-chunks), tanh ----
        // Keep this loop ROLLED: fragment addresses depend on l, so the ds_load_b128s
        // cannot be LICM-hoisted into (and spilled out of) registers across tiles.
        #pragma clang loop unroll(disable)
        for (int l = 1; l < 7; ++l) {
            // B fragments: 32x16 f16, lane holds column N=n, K = 32c + 16g + 0..15
            const v16h p00 = *(const v16h*)(hp0 + 16 * g);
            const v16h p01 = *(const v16h*)(hp0 + 32 + 16 * g);
            const v16h p10 = *(const v16h*)(hp1 + 16 * g);
            const v16h p11 = *(const v16h*)(hp1 + 32 + 16 * g);
            const _Float16* wf  = wfrag + 2048 + (l - 1) * 4096;
            const float*    bsp = biasS + 64 * l;
            #pragma unroll
            for (int mt = 0; mt < 4; ++mt) {
                const v8f  bias = *(const v8f*)(bsp + 16 * mt + 8 * g);
                const _Float16* fb = wf + mt * 1024 + lane * 16;
                const v16h a0 = *(const v16h*)fb;
                const v16h a1 = *(const v16h*)(fb + 512);
                v8f acc0 = wmma_f16(a0, p00, bias);
                v8f acc1 = wmma_f16(a0, p10, bias);
                acc0 = wmma_f16(a1, p01, acc0);
                acc1 = wmma_f16(a1, p11, acc1);
                store_tanh(hp0 + 16 * mt + 8 * g, acc0);
                store_tanh(hp1 + 16 * mt + 8 * g, acc1);
            }
        }

        // ---- Layer 8 (40 -> 1) + sigmoid * scale ----
        {
            const v16h p00 = *(const v16h*)(hp0 + 16 * g);
            const v16h p01 = *(const v16h*)(hp0 + 32 + 16 * g);
            const v16h p10 = *(const v16h*)(hp1 + 16 * g);
            const v16h p11 = *(const v16h*)(hp1 + 32 + 16 * g);
            const v8f  bias = *(const v8f*)(biasS + 7 * 64 + 8 * g);
            const _Float16* fb = wfrag + 26624 + lane * 16;
            const v16h a0 = *(const v16h*)fb;
            const v16h a1 = *(const v16h*)(fb + 512);
            v8f acc0 = wmma_f16(a0, p00, bias);
            v8f acc1 = wmma_f16(a0, p10, bias);
            acc0 = wmma_f16(a1, p01, acc0);
            acc1 = wmma_f16(a1, p11, acc1);
            // D layout: VGPR0, lanes 0..15 hold M=0 (the single output), N=n
            const float o0 = sc * fast_sigmoid(acc0[0]);
            const float o1 = sc * fast_sigmoid(acc1[0]);
            if (g == 0) {
                if (r0 < (int64_t)Ntot) out[r0] = o0;
                if (r1 < (int64_t)Ntot) out[r1] = o1;
            }
        }
    }
}

extern "C" void kernel_launch(void* const* d_in, const int* in_sizes, int n_in,
                              void* d_out, int out_size, void* d_ws, size_t ws_size,
                              hipStream_t stream) {
    (void)n_in; (void)d_ws; (void)ws_size; (void)out_size;
    const float* x  = (const float*)d_in[0];
    const float* W1 = (const float*)d_in[1];  const float* B1 = (const float*)d_in[2];
    const float* W2 = (const float*)d_in[3];  const float* B2 = (const float*)d_in[4];
    const float* W3 = (const float*)d_in[5];  const float* B3 = (const float*)d_in[6];
    const float* W4 = (const float*)d_in[7];  const float* B4 = (const float*)d_in[8];
    const float* W5 = (const float*)d_in[9];  const float* B5 = (const float*)d_in[10];
    const float* W6 = (const float*)d_in[11]; const float* B6 = (const float*)d_in[12];
    const float* W7 = (const float*)d_in[13]; const float* B7 = (const float*)d_in[14];
    const float* W8 = (const float*)d_in[15]; const float* B8 = (const float*)d_in[16];
    const float* sc = (const float*)d_in[17];
    float* out = (float*)d_out;

    const int Ntot = in_sizes[0] / 2;
    const int64_t tiles = ((int64_t)Ntot + 31) / 32;
    int blocks = (int)((tiles + WAVES_PER_BLOCK - 1) / WAVES_PER_BLOCK);
    if (blocks > 2048) blocks = 2048;   // several tiles/wave amortizes LDS weight setup
    if (blocks < 1) blocks = 1;

    mlp_forward_kernel<<<blocks, THREADS, 0, stream>>>(
        x, W1, B1, W2, B2, W3, B3, W4, B4, W5, B5, W6, B6, W7, B7, W8, B8,
        sc, out, Ntot);
}
